// CLGODE_4612794876707
// MI455X (gfx1250) — matrix operand order repaired and checked by reading, not compiled, not run on hardware
//
#include <hip/hip_runtime.h>
#include <hip/hip_bf16.h>
#include <math.h>

typedef __attribute__((ext_vector_type(16))) _Float16 v16h;
typedef __attribute__((ext_vector_type(8)))  _Float16 v8h;
typedef __attribute__((ext_vector_type(8)))  float    v8f;
typedef __attribute__((ext_vector_type(4)))  unsigned int u32x4;
typedef __attribute__((ext_vector_type(8)))  int      i32x8;
typedef __attribute__((ext_vector_type(4)))  int      i32x4;

#define B_     16
#define N_     400
#define MD_    8
#define L_     64
#define H_     128
#define T_     8
#define STEPS_ 8
#define COVD_  18

// ---- output offsets (floats) in d_out, reference tuple order ----
#define OFF_XHAT   0
#define OFF_LOGIT  409600
#define OFF_W1     20889600
#define OFF_ZM     41369600
#define OFF_ZC     44646400
#define OFF_MUM    47923200
#define OFF_LVM    48332800
#define OFF_MUC    48742400
#define OFF_LVC    49152000
#define OFF_W2     49561600
#define OFF_LNEW   70041600

// ---- workspace byte offsets ----
#define WS_COV    0          // 288 f
#define WS_DIS    1280       // 6400 f
#define WS_YV     26880      // 51200 f
#define WS_HX     231680     // 51200 f
#define WS_Z0     436480     // 819200 f
#define WS_WPACK  3713280    // 116224 B  (W1pack 81920 + W2pack 32768 + b1 1024 + b2 512)
#define WS_ZC16   3829504    // 3276800 halfs

#define WPACK_BYTES 116224
#define LDS_WAVE_SCRATCH 13312   // feat 2560 halfs (5120B) + hid 4096 halfs (8192B)
#define LDS_TOTAL (WPACK_BYTES + 8*LDS_WAVE_SCRATCH)   // 222720 B < 320KB

__device__ __forceinline__ v8f wmma16(v16h a, v16h b, v8f c) {
  return __builtin_amdgcn_wmma_f32_16x16x32_f16(false, a, false, b, (short)0, c,
                                                false, false);
}

// ------------------------------------------------------------------
// One ODE MLP evaluation for a 16-row tile held as 8 C-layout v8f tiles.
// Scatters y = z + coef*kin into LDS A-fragment layout, runs
// feat(16x160) @ W1cat(160x256) -> relu -> blockdiag W2 -> kout (8 tiles).
// kin and kout may alias: kin is fully read before kout is written.
// ------------------------------------------------------------------
__device__ __forceinline__ void ode_eval(const v8f* y, const v8f* kin, float coef,
                                         float tstage,
                                         _Float16* feat, _Float16* hid,
                                         const _Float16* lW1, const _Float16* lW2,
                                         const float* lb1, const float* lb2,
                                         int lane, v8f* kout)
{
  const int col16 = lane & 15;
  const int rowhi = (lane >> 4) << 3;       // C-layout: rows r or r+8
  const int lh16  = ((lane >> 3) & 1) << 4; // A-layout lane-half select

  // scatter z part (feat cols 0..127 -> ktiles 0..3)
  #pragma unroll
  for (int c = 0; c < 8; ++c) {
    const int kt = c >> 1;
    const int e  = (lane & 7) + ((c & 1) << 3);
    #pragma unroll
    for (int r = 0; r < 8; ++r) {
      const int lanep = (r + rowhi) + lh16;
      float v = y[c][r] + coef * kin[c][r];
      feat[(kt * 32 + lanep) * 16 + e] = (_Float16)v;
    }
  }
  // time slot: feat col 146 -> o=18 -> lanes 0..15, e=10
  if (lane < 16) feat[(4 * 32 + lane) * 16 + 10] = (_Float16)tstage;

  // gather A fragments (contiguous 16 halfs per lane per ktile)
  v16h af[5];
  #pragma unroll
  for (int kt = 0; kt < 5; ++kt)
    af[kt] = *(const v16h*)(feat + (kt * 32 + lane) * 16);

  // GEMM1: hidden = relu(feat @ W1cat + b1cat); scatter to hid A-frag layout
  #pragma unroll
  for (int nt = 0; nt < 16; ++nt) {
    float bv = lb1[nt * 16 + col16];
    v8f acc = {bv, bv, bv, bv, bv, bv, bv, bv};
    #pragma unroll
    for (int kt = 0; kt < 5; ++kt) {
      v16h bf = *(const v16h*)(lW1 + ((kt * 16 + nt) * 32 + lane) * 16);
      acc = wmma16(af[kt], bf, acc);
    }
    const int kth = nt >> 1;
    const int e   = (lane & 7) + ((nt & 1) << 3);
    #pragma unroll
    for (int r = 0; r < 8; ++r) {
      const int lanep = (r + rowhi) + lh16;
      float v = fmaxf(acc[r], 0.f);
      hid[(kth * 32 + lanep) * 16 + e] = (_Float16)v;
    }
  }

  // GEMM2: two block-diagonal 128x64 nets
  #pragma unroll
  for (int net = 0; net < 2; ++net) {
    v16h ha[4];
    #pragma unroll
    for (int kt = 0; kt < 4; ++kt)
      ha[kt] = *(const v16h*)(hid + ((net * 4 + kt) * 32 + lane) * 16);
    #pragma unroll
    for (int nt = 0; nt < 4; ++nt) {
      float bv = lb2[net * 64 + nt * 16 + col16];
      v8f acc = {bv, bv, bv, bv, bv, bv, bv, bv};
      #pragma unroll
      for (int kt = 0; kt < 4; ++kt) {
        v16h bf = *(const v16h*)(lW2 + (((net * 4 + kt) * 4 + nt) * 32 + lane) * 16);
        acc = wmma16(ha[kt], bf, acc);
      }
      kout[net * 4 + nt] = acc;
    }
  }
}

// ------------------------------------------------------------------
// Persistent RK4 integrator: 50 blocks x 8 waves, 16 rows per wave.
// Weights staged LDS-resident via the Tensor Data Mover when available.
// ------------------------------------------------------------------
__global__ void __launch_bounds__(256)
ode_kernel(const float* __restrict__ z0g, const float* __restrict__ times,
           const float* __restrict__ cov, const float* __restrict__ wpack,
           float* __restrict__ zm_out, float* __restrict__ zc_out,
           _Float16* __restrict__ zc16)
{
  extern __shared__ char smem[];
  _Float16* sW1 = (_Float16*)smem;                 // 40960 halfs
  _Float16* sW2 = sW1 + 40960;                     // 16384 halfs
  const float* sb1 = (const float*)(smem + 114688);
  const float* sb2 = (const float*)(smem + 115712);

  const int lane = threadIdx.x & 31;
  const int wave = threadIdx.x >> 5;

#if defined(__HIP_DEVICE_COMPILE__) && __has_builtin(__builtin_amdgcn_tensor_load_to_lds)
  // Single TDM descriptor: 2D tensor, one row of 14528 8-byte elements
  // = the entire 113.5KB packed weight block, global -> LDS offset 0.
  if (wave == 0) {
    unsigned long ga = (unsigned long)(const void*)wpack;
    const unsigned elems = WPACK_BYTES / 8;        // data_size = 8 bytes
    u32x4 g0 = {0u, 0u, 0u, 0u};
    g0[0] = 1u;                                    // count=1 (valid user D#)
    g0[1] = 0u;                                    // lds_addr = 0 bytes
    g0[2] = (unsigned)(ga & 0xffffffffu);          // global_addr[31:0]
    g0[3] = (unsigned)((ga >> 32) & 0x01ffffffu) | 0x80000000u; // addr[56:32] | type=2
    i32x8 g1 = {0, 0, 0, 0, 0, 0, 0, 0};
    g1[0] = (int)(3u << 16);                       // wg_mask=0, data_size=3 (8B)
    g1[1] = (int)((elems & 0xffffu) << 16);        // tensor_dim0[15:0]
    g1[2] = (int)(((elems >> 16) & 0xffffu) | (1u << 16)); // dim0 hi | tensor_dim1=1
    g1[3] = (int)((elems & 0xffffu) << 16);        // tile_dim0 = elems
    g1[4] = 1;                                     // tile_dim1 = 1, tile_dim2 = 0
    g1[5] = (int)elems;                            // tensor_dim0_stride[31:0]
    i32x4 gz4 = {0, 0, 0, 0};
    i32x8 gz8 = {0, 0, 0, 0, 0, 0, 0, 0};
    __builtin_amdgcn_tensor_load_to_lds(g0, g1, gz4, gz4, gz8, 0);
    __builtin_amdgcn_s_wait_tensorcnt(0);
  }
#else
  {
    const float4* src = (const float4*)wpack;
    float4* dst = (float4*)smem;
    for (int i = threadIdx.x; i < WPACK_BYTES / 16; i += 256) dst[i] = src[i];
  }
#endif
  __syncthreads();

  _Float16* feat = (_Float16*)(smem + WPACK_BYTES + wave * LDS_WAVE_SCRATCH);
  _Float16* hid  = feat + 2560;

  const int tile = blockIdx.x * 8 + wave;   // 0..399
  const int b    = tile / 25;
  const int n0   = (tile % 25) * 16;

  const int col16 = lane & 15;
  const int rowhi = (lane >> 4) << 3;

  // static feat region kt=4: cov cols 128..145, t col 146 (filled per eval), pad 0
  #pragma unroll
  for (int e = 0; e < 16; ++e) {
    int o = (e & 7) | (((lane >> 4) & 1) << 3) | (((e >> 3) & 1) << 4);
    float v = (o < COVD_) ? cov[b * COVD_ + o] : 0.f;
    feat[(4 * 32 + lane) * 16 + e] = (_Float16)v;
  }

  // load z0 rows into C-layout tiles
  v8f z[8];
  #pragma unroll
  for (int c = 0; c < 8; ++c)
    #pragma unroll
    for (int r = 0; r < 8; ++r)
      z[c][r] = z0g[((long)(b * N_ + n0 + r + rowhi)) * 128 + c * 16 + col16];

  auto write_out = [&](int tidx) {
    #pragma unroll
    for (int c = 0; c < 8; ++c) {
      #pragma unroll
      for (int r = 0; r < 8; ++r) {
        int n = n0 + r + rowhi;
        long base = (((long)(b * T_ + tidx)) * N_ + n) * 64 + (c & 3) * 16 + col16;
        float v = z[c][r];
        if (c < 4) {
          zm_out[base] = v;
        } else {
          zc_out[base] = v;
          zc16[base]   = (_Float16)v;
        }
      }
    }
  };
  write_out(0);

  v8f k[8], ksum[8];
  for (int idx = 1; idx < T_; ++idx) {
    float t0 = times[b * T_ + idx - 1];
    float t1 = times[b * T_ + idx];
    float h  = (t1 - t0) / (float)STEPS_;
    float t  = t0;
    for (int s = 0; s < STEPS_; ++s) {
      ode_eval(z, z, 0.f, t, feat, hid, sW1, sW2, sb1, sb2, lane, k);
      #pragma unroll
      for (int c = 0; c < 8; ++c) ksum[c] = k[c];
      ode_eval(z, k, 0.5f * h, t + 0.5f * h, feat, hid, sW1, sW2, sb1, sb2, lane, k);
      #pragma unroll
      for (int c = 0; c < 8; ++c) ksum[c] += 2.f * k[c];
      ode_eval(z, k, 0.5f * h, t + 0.5f * h, feat, hid, sW1, sW2, sb1, sb2, lane, k);
      #pragma unroll
      for (int c = 0; c < 8; ++c) ksum[c] += 2.f * k[c];
      ode_eval(z, k, h, t + h, feat, hid, sW1, sW2, sb1, sb2, lane, k);
      #pragma unroll
      for (int c = 0; c < 8; ++c) {
        ksum[c] += k[c];
        z[c] += (h / 6.f) * ksum[c];
      }
      t += h;
    }
    write_out(idx);
  }
}

// ------------------------------------------------------------------
// score = zc zc^T per (b,t); one wave per 16x16 tile; fused elementwise outs.
// ------------------------------------------------------------------
__global__ void __launch_bounds__(256)
score_kernel(const _Float16* __restrict__ zc16, float* __restrict__ out,
             const float* alpha, const float* delta, const float* gammap,
             const float* betap, const float* alpha_new, const float* delta_new)
{
  const int lane = threadIdx.x & 31;
  const int wave = threadIdx.x >> 5;
  const int tileId = blockIdx.x * 8 + wave;     // 0..79999
  const int bt  = tileId / 625;
  const int rem = tileId % 625;
  const int ti = rem / 25, tj = rem % 25;

  const _Float16* zrow = zc16 + (long)bt * N_ * 64;

  // A fragments: rows ti*16.., K=64 split into two ktiles
  const int arow  = ti * 16 + (lane & 15);
  const int base0 = ((lane >> 4) & 1) * 16;
  v16h a0, a1;
  {
    v8h lo0 = *(const v8h*)(zrow + (long)arow * 64 + base0);
    v8h hi0 = *(const v8h*)(zrow + (long)arow * 64 + base0 + 8);
    v8h lo1 = *(const v8h*)(zrow + (long)arow * 64 + 32 + base0);
    v8h hi1 = *(const v8h*)(zrow + (long)arow * 64 + 32 + base0 + 8);
    #pragma unroll
    for (int e = 0; e < 8; ++e) {
      a0[e] = lo0[e]; a0[8 + e] = hi0[e];
      a1[e] = lo1[e]; a1[8 + e] = hi1[e];
    }
  }
  // B fragments: columns tj*16.. (B[k][m] = zc[m][k] -> contiguous per lane)
  const int bcol  = tj * 16 + (lane & 15);
  const int bhalf = ((lane >> 4) & 1) * 16;
  v16h b0 = *(const v16h*)(zrow + (long)bcol * 64 + bhalf);
  v16h b1 = *(const v16h*)(zrow + (long)bcol * 64 + 32 + bhalf);

  v8f s = {0.f, 0.f, 0.f, 0.f, 0.f, 0.f, 0.f, 0.f};
  s = wmma16(a0, b0, s);
  s = wmma16(a1, b1, s);

  const float al = *alpha, de = *delta, ga = *gammap, be = *betap;
  const float aln = *alpha_new, den = *delta_new;
  #pragma unroll
  for (int r = 0; r < 8; ++r) {
    int row = ti * 16 + r + ((lane >> 4) << 3);
    int col = tj * 16 + (lane & 15);
    long base = (long)bt * (N_ * N_) + (long)row * N_ + col;
    float sv = s[r];
    float x  = ga * sv + be;
    float sp = (x > 20.f) ? x : log1pf(__expf(x));
    float w  = (row == col) ? 0.f : sp;
    out[OFF_LOGIT + base] = al * (sv - de);
    out[OFF_W1 + base]    = w;
    out[OFF_W2 + base]    = w;
    out[OFF_LNEW + base]  = aln * (sv - den);
  }
}

// ------------------------------------------------------------------
// small setup / elementwise kernels
// ------------------------------------------------------------------
__global__ void cov_kernel(const int* __restrict__ sex, const int* __restrict__ site,
                           const float* __restrict__ covr,
                           const float* __restrict__ sex_emb,
                           const float* __restrict__ site_emb,
                           float* __restrict__ cov)
{
  int t = threadIdx.x;
  if (t >= B_ * COVD_) return;
  int b = t / COVD_, o = t % COVD_;
  float v;
  if (o < 8)       v = sex_emb[sex[b] * 8 + o];
  else if (o < 16) v = site_emb[site[b] * 8 + (o - 8)];
  else             v = covr[b * 2 + (o - 16)];
  cov[t] = v;
}

__global__ void pack_kernel(const float* __restrict__ om_W1, const float* __restrict__ oc_W1,
                            const float* __restrict__ om_W2, const float* __restrict__ oc_W2,
                            const float* __restrict__ om_b1, const float* __restrict__ oc_b1,
                            const float* __restrict__ om_b2, const float* __restrict__ oc_b2,
                            char* __restrict__ wpack)
{
  _Float16* W1p = (_Float16*)wpack;
  _Float16* W2p = W1p + 40960;
  float* b1c = (float*)(wpack + 114688);
  float* b2c = (float*)(wpack + 115712);
  int q = blockIdx.x * blockDim.x + threadIdx.x;
  if (q < 40960) {                       // W1cat (160x256) -> B-frag order
    int e = q & 15, lane = (q >> 4) & 31, lin = q >> 9;
    int kt = lin >> 4, nt = lin & 15;
    int k = kt * 32 + ((e >> 3) & 1) * 16 + ((lane >> 4) & 1) * 8 + (e & 7);   // feat dim 0..159
    int j = nt * 16 + (lane & 15);                  // hidden col 0..255
    float v = 0.f;
    if (k < 147) v = (j < 128) ? om_W1[k * 128 + j] : oc_W1[k * 128 + (j - 128)];
    W1p[q] = (_Float16)v;
  } else if (q < 57344) {                // W2 (2 nets, 128x64) -> B-frag order
    int q2 = q - 40960;
    int e = q2 & 15, lane = (q2 >> 4) & 31, lin = q2 >> 9;
    int nt = lin & 3, kt = (lin >> 2) & 3, net = lin >> 4;
    int k = kt * 32 + ((e >> 3) & 1) * 16 + ((lane >> 4) & 1) * 8 + (e & 7);
    int j = nt * 16 + (lane & 15);
    W2p[q2] = (_Float16)(net == 0 ? om_W2[k * 64 + j] : oc_W2[k * 64 + j]);
  } else if (q < 57344 + 256) {
    int j = q - 57344;
    b1c[j] = (j < 128) ? om_b1[j] : oc_b1[j - 128];
  } else if (q < 57344 + 384) {
    int j = q - 57344 - 256;
    b2c[j] = (j < 64) ? om_b2[j] : oc_b2[j - 64];
  }
}

__global__ void deg_kernel(const float* __restrict__ a0, const float* __restrict__ x0,
                           float* __restrict__ dis, float* __restrict__ yv)
{
  int idx = blockIdx.x * blockDim.x + threadIdx.x;
  if (idx >= B_ * N_) return;
  int b = idx / N_, n = idx % N_;
  const float* arow = a0 + ((long)b * N_ + n) * N_;
  float sum = 1.0f;                      // identity diagonal of a_hat
  for (int m = 0; m < N_; ++m) sum += arow[m];
  float d = rsqrtf(sum + 1e-8f);
  dis[idx] = d;
  #pragma unroll
  for (int k = 0; k < 8; ++k) yv[idx * 8 + k] = d * x0[idx * 8 + k];
}

__global__ void hx_kernel(const float* __restrict__ a0, const float* __restrict__ dis,
                          const float* __restrict__ yv, float* __restrict__ hx)
{
  int idx = blockIdx.x * blockDim.x + threadIdx.x;   // (b,n,d)
  if (idx >= B_ * N_ * 8) return;
  int d = idx & 7;
  int bn = idx >> 3;
  int b = bn / N_, n = bn % N_;
  const float* arow = a0 + ((long)b * N_ + n) * N_;
  const float* yb = yv + (long)b * N_ * 8;
  float acc = yb[n * 8 + d];             // identity term
  for (int m = 0; m < N_; ++m) acc += arow[m] * yb[m * 8 + d];
  hx[idx] = dis[bn] * acc;
}

__global__ void __launch_bounds__(128)
enc_kernel(const float* __restrict__ hx,
           const float* em_W1, const float* em_b1, const float* em_Wmu,
           const float* em_bmu, const float* em_Wlv, const float* em_blv,
           const float* ec_W1, const float* ec_b1, const float* ec_Wmu,
           const float* ec_bmu, const float* ec_Wlv, const float* ec_blv,
           float* __restrict__ out, float* __restrict__ z0)
{
  __shared__ float sx[8];
  __shared__ float hm[128], hc[128];
  long row = blockIdx.x;                 // 0..6399
  int j = threadIdx.x;
  if (j < 8) sx[j] = hx[row * 8 + j];
  __syncthreads();
  float am = em_b1[j], ac = ec_b1[j];
  #pragma unroll
  for (int k = 0; k < 8; ++k) {
    am += sx[k] * em_W1[k * 128 + j];
    ac += sx[k] * ec_W1[k * 128 + j];
  }
  hm[j] = fmaxf(am, 0.f);
  hc[j] = fmaxf(ac, 0.f);
  __syncthreads();
  if (j < 64) {
    float mm = em_bmu[j], mc = ec_bmu[j];
    for (int k = 0; k < 128; ++k) {
      mm += hm[k] * em_Wmu[k * 64 + j];
      mc += hc[k] * ec_Wmu[k * 64 + j];
    }
    out[OFF_MUM + row * 64 + j] = mm;
    out[OFF_MUC + row * 64 + j] = mc;
    z0[row * 128 + j]      = mm;
    z0[row * 128 + 64 + j] = mc;
  } else {
    int jj = j - 64;
    float lm = em_blv[jj], lc = ec_blv[jj];
    for (int k = 0; k < 128; ++k) {
      lm += hm[k] * em_Wlv[k * 64 + jj];
      lc += hc[k] * ec_Wlv[k * 64 + jj];
    }
    out[OFF_LVM + row * 64 + jj] = lm;
    out[OFF_LVC + row * 64 + jj] = lc;
  }
}

__global__ void __launch_bounds__(128)
dec_kernel(const float* __restrict__ zm, const float* dm_W1, const float* dm_b1,
           const float* dm_W2, const float* dm_b2, float* __restrict__ xhat)
{
  __shared__ float sz[64];
  __shared__ float hd[128];
  long row = blockIdx.x;                 // 0..51199
  int j = threadIdx.x;
  if (j < 64) sz[j] = zm[row * 64 + j];
  __syncthreads();
  float a = dm_b1[j];
  #pragma unroll 8
  for (int k = 0; k < 64; ++k) a += sz[k] * dm_W1[k * 128 + j];
  hd[j] = fmaxf(a, 0.f);
  __syncthreads();
  if (j < 8) {
    float o = dm_b2[j];
    for (int k = 0; k < 128; ++k) o += hd[k] * dm_W2[k * 8 + j];
    xhat[row * 8 + j] = o;
  }
}

// ------------------------------------------------------------------
extern "C" void kernel_launch(void* const* d_in, const int* in_sizes, int n_in,
                              void* d_out, int out_size, void* d_ws, size_t ws_size,
                              hipStream_t stream)
{
  const float* a0    = (const float*)d_in[0];
  const float* x0    = (const float*)d_in[1];
  const float* times = (const float*)d_in[2];
  const int*   sex   = (const int*)d_in[3];
  const int*   site  = (const int*)d_in[4];
  const float* covr  = (const float*)d_in[5];
  const float* em_W1 = (const float*)d_in[6];
  const float* em_b1 = (const float*)d_in[7];
  const float* em_Wmu= (const float*)d_in[8];
  const float* em_bmu= (const float*)d_in[9];
  const float* em_Wlv= (const float*)d_in[10];
  const float* em_blv= (const float*)d_in[11];
  const float* ec_W1 = (const float*)d_in[12];
  const float* ec_b1 = (const float*)d_in[13];
  const float* ec_Wmu= (const float*)d_in[14];
  const float* ec_bmu= (const float*)d_in[15];
  const float* ec_Wlv= (const float*)d_in[16];
  const float* ec_blv= (const float*)d_in[17];
  const float* sex_emb = (const float*)d_in[18];
  const float* site_emb= (const float*)d_in[19];
  const float* om_W1 = (const float*)d_in[20];
  const float* om_b1 = (const float*)d_in[21];
  const float* om_W2 = (const float*)d_in[22];
  const float* om_b2 = (const float*)d_in[23];
  const float* oc_W1 = (const float*)d_in[24];
  const float* oc_b1 = (const float*)d_in[25];
  const float* oc_W2 = (const float*)d_in[26];
  const float* oc_b2 = (const float*)d_in[27];
  const float* dm_W1 = (const float*)d_in[28];
  const float* dm_b1 = (const float*)d_in[29];
  const float* dm_W2 = (const float*)d_in[30];
  const float* dm_b2 = (const float*)d_in[31];
  const float* alpha     = (const float*)d_in[32];
  const float* delta     = (const float*)d_in[33];
  const float* gammap    = (const float*)d_in[34];
  const float* betap     = (const float*)d_in[35];
  const float* alpha_new = (const float*)d_in[36];
  const float* delta_new = (const float*)d_in[37];

  char* ws = (char*)d_ws;
  float*    covb  = (float*)(ws + WS_COV);
  float*    dis   = (float*)(ws + WS_DIS);
  float*    yv    = (float*)(ws + WS_YV);
  float*    hx    = (float*)(ws + WS_HX);
  float*    z0    = (float*)(ws + WS_Z0);
  float*    wpack = (float*)(ws + WS_WPACK);
  _Float16* zc16  = (_Float16*)(ws + WS_ZC16);

  float* out = (float*)d_out;

  cov_kernel<<<1, 288, 0, stream>>>(sex, site, covr, sex_emb, site_emb, covb);
  pack_kernel<<<226, 256, 0, stream>>>(om_W1, oc_W1, om_W2, oc_W2,
                                       om_b1, oc_b1, om_b2, oc_b2, (char*)wpack);
  deg_kernel<<<25, 256, 0, stream>>>(a0, x0, dis, yv);
  hx_kernel<<<200, 256, 0, stream>>>(a0, dis, yv, hx);
  enc_kernel<<<6400, 128, 0, stream>>>(hx, em_W1, em_b1, em_Wmu, em_bmu, em_Wlv, em_blv,
                                       ec_W1, ec_b1, ec_Wmu, ec_bmu, ec_Wlv, ec_blv,
                                       out, z0);
  ode_kernel<<<50, 256, LDS_TOTAL, stream>>>(z0, times, covb, wpack,
                                             out + OFF_ZM, out + OFF_ZC, zc16);
  dec_kernel<<<51200, 128, 0, stream>>>(out + OFF_ZM, dm_W1, dm_b1, dm_W2, dm_b2,
                                        out + OFF_XHAT);
  score_kernel<<<10000, 256, 0, stream>>>(zc16, out, alpha, delta, gammap, betap,
                                          alpha_new, delta_new);
}
